// GumbelMoE_36893769072967
// MI455X (gfx1250) — matrix operand order repaired and checked by reading, not compiled, not run on hardware
//
#include <hip/hip_runtime.h>
#include <hip/hip_bf16.h>
#include <math.h>

#ifndef __has_builtin
#define __has_builtin(x) 0
#endif

#if __has_builtin(__builtin_amdgcn_tensor_load_to_lds) && __has_builtin(__builtin_amdgcn_s_wait_tensorcnt)
#define HAVE_TDM 1
#else
#define HAVE_TDM 0
#endif

// ---------------------------------------------------------------------------
// GumbelMoE forward for MI455X (gfx1250, wave32, WMMA + TDM)
//   x_s = concat(attn1,attn2) @ W_share + b_share        (bf16 WMMA, f32 acc)
//   p = onehot(argmax(x_s @ W_router + b + gumbel)); out row selected by p
// ---------------------------------------------------------------------------

typedef __bf16 bf16;
typedef __attribute__((ext_vector_type(4)))  bf16  bf16x4;
typedef __attribute__((ext_vector_type(8)))  bf16  bf16x8;
typedef __attribute__((ext_vector_type(16))) bf16  v16bf;
typedef __attribute__((ext_vector_type(8)))  float v8f;
typedef __attribute__((ext_vector_type(4)))  unsigned int u32x4;
typedef __attribute__((ext_vector_type(4)))  int   i32x4;
typedef __attribute__((ext_vector_type(8)))  int   i32x8;

#define D_MODEL   2048
#define K_TOTAL   4096
#define N_TOKENS  16384   // B*S = 4*4096
#define KT_STEPS  128     // K_TOTAL / 32

// ---------------------------------------------------------------------------
// W_share pre-pass: f32 [4096 k][2048 n] -> bf16 Wt [2048 n][4096 k].
// Done once per launch (48 MB of traffic ~ 2 us); W is then consumed 128x
// from L2 without per-block conversion or transposition.
// ---------------------------------------------------------------------------
__global__ __launch_bounds__(256)
void transpose_ws_kernel(const float* __restrict__ Ws, bf16* __restrict__ Wt)
{
    __shared__ bf16 tile[32][33];
    const int tid = threadIdx.x;
    const int k0 = blockIdx.x * 32;     // 128 blocks
    const int n0 = blockIdx.y * 32;     // 64 blocks
    {
        const int r = tid >> 3;             // k_local 0..31
        const int c = (tid & 7) * 4;        // n_local 0,4..28
        const float4 v = *reinterpret_cast<const float4*>(
            Ws + (size_t)(k0 + r) * D_MODEL + n0 + c);
        tile[c + 0][r] = (bf16)v.x;
        tile[c + 1][r] = (bf16)v.y;
        tile[c + 2][r] = (bf16)v.z;
        tile[c + 3][r] = (bf16)v.w;
    }
    __syncthreads();
    {
        const int rn = tid >> 3;            // n_local
        const int ck = (tid & 7) * 4;       // k_local
        bf16x4 o = { tile[rn][ck + 0], tile[rn][ck + 1],
                     tile[rn][ck + 2], tile[rn][ck + 3] };
        *reinterpret_cast<bf16x4*>(Wt + (size_t)(n0 + rn) * K_TOTAL + k0 + ck) = o;
    }
}

// ---------------------------------------------------------------------------
// TDM descriptor: copy a 32(k) x 128(n) bf16 tile of Wt into LDS, with
// hardware padding 16 DWORDs data + 4 DWORDs pad = 80 B LDS row stride
// (matches Bs[n][40] layout). 2D tensor: dim0 = k (contiguous), dim1 = n.
// ---------------------------------------------------------------------------
#if HAVE_TDM
__device__ __forceinline__ void tdm_load_B(const bf16* gsrc, unsigned lds_off)
{
    const unsigned long long ga = (unsigned long long)(size_t)gsrc;
    u32x4 g0;
    g0[0] = 1u;                                       // count=1, user mode
    g0[1] = lds_off;                                  // lds_addr
    g0[2] = (unsigned)ga;                             // global_addr[31:0]
    g0[3] = (unsigned)((ga >> 32) & 0x1FFFFFFu) | (2u << 30);  // addr[56:32], type=2
    i32x8 g1;
    g1[0] = (int)((1u << 16) |                        // data_size = 2B
                  (1u << 20) |                        // pad_enable
                  (3u << 22) |                        // pad_interval: 16 DWORDs
                  (3u << 25));                        // pad_amount: 4 DWORDs
    g1[1] = (int)(((unsigned)K_TOTAL & 0xFFFFu) << 16);         // tensor_dim0 lo16
    g1[2] = (int)((((unsigned)K_TOTAL >> 16) & 0xFFFFu) |
                  (((unsigned)D_MODEL & 0xFFFFu) << 16));       // dim0 hi, dim1 lo
    g1[3] = (int)((((unsigned)D_MODEL >> 16) & 0xFFFFu) |
                  (32u << 16));                                 // dim1 hi, tile_dim0=32
    g1[4] = (int)128u;                                          // tile_dim1=128, tile_dim2=0
    g1[5] = (int)(unsigned)K_TOTAL;                             // dim0_stride lo32 (elems)
    g1[6] = 0;                                                  // stride hi, dim1_stride lo
    g1[7] = 0;
    const i32x4 z4 = {0, 0, 0, 0};
#if __clang_major__ >= 23
    const i32x8 z8 = {0, 0, 0, 0, 0, 0, 0, 0};
    __builtin_amdgcn_tensor_load_to_lds(g0, g1, z4, z4, z8, 0);
#else
    __builtin_amdgcn_tensor_load_to_lds(g0, g1, z4, z4, 0);
#endif
}
#endif

// ---------------------------------------------------------------------------
// Kernel 1: x_s GEMM. Block tile 128(M) x 128(N), K-step 32, double-buffered
// LDS. 8 waves: 2(M) x 4(N); each wave -> 4x2 WMMA tiles (64x32).
// BPATH: 0 = B via TDM from Wt, 1 = B via vector loads from Wt, 2 = legacy f32.
// ---------------------------------------------------------------------------
template <int BPATH>
__global__ __launch_bounds__(256)
void gemm_xs_kernel(const float* __restrict__ attn1,
                    const float* __restrict__ attn2,
                    const float* __restrict__ Ws,      // [4096, 2048] f32
                    const bf16*  __restrict__ Wt,      // [2048, 4096] bf16 (transposed)
                    const float* __restrict__ bsh,     // [2048]
                    float* __restrict__ xs_out)        // [16384, 2048]
{
    __shared__ __align__(16) bf16 As[2][128][40];   // [m][k]  20 KB
    __shared__ __align__(16) bf16 Bs[2][128][40];   // [n][k]  20 KB

    const int tid  = threadIdx.x;
    const int lane = tid & 31;
    const int wave = tid >> 5;
    const int gm0  = blockIdx.y * 128;
    const int gn0  = blockIdx.x * 128;
    const int wm   = (wave >> 2) * 64;
    const int wn   = (wave & 3) * 32;

    v8f acc[4][2] = {};
    float4 ar[4];          // A staging regs
    float4 brf[4];         // B staging regs (legacy path)
    bf16x8 brh[2];         // B staging regs (Wt vector path)

    auto load_A = [&](int kt) {
        const int kg = kt * 32;
        const float* asrc = (kg < 2048) ? attn1 : attn2;   // concat handled here
        const int kc = kg & 2047;
#pragma unroll
        for (int i = 0; i < 4; ++i) {
            const int idx = i * 256 + tid;                 // 0..1023
            ar[i] = *reinterpret_cast<const float4*>(
                asrc + (size_t)(gm0 + (idx >> 3)) * D_MODEL + kc + (idx & 7) * 4);
        }
    };
    auto store_A = [&](int buf) {
#pragma unroll
        for (int i = 0; i < 4; ++i) {
            const int idx = i * 256 + tid;
            const int row = idx >> 3, c = (idx & 7) * 4;
            bf16x4 a4 = { (bf16)ar[i].x, (bf16)ar[i].y, (bf16)ar[i].z, (bf16)ar[i].w };
            *reinterpret_cast<bf16x4*>(&As[buf][row][c]) = a4;
        }
    };

    auto load_B = [&](int kt) {
        const int kg = kt * 32;
        if constexpr (BPATH == 1) {
#pragma unroll
            for (int i = 0; i < 2; ++i) {
                const int idx = i * 256 + tid;             // 0..511
                const int row = idx >> 2, c8 = (idx & 3) * 8;
                brh[i] = *reinterpret_cast<const bf16x8*>(
                    Wt + (size_t)(gn0 + row) * K_TOTAL + kg + c8);
            }
        } else if constexpr (BPATH == 2) {
#pragma unroll
            for (int i = 0; i < 4; ++i) {
                const int idx = i * 256 + tid;
                brf[i] = *reinterpret_cast<const float4*>(
                    Ws + (size_t)(kg + (idx >> 5)) * D_MODEL + gn0 + (idx & 31) * 4);
            }
        }
    };
    auto store_B = [&](int buf) {
        if constexpr (BPATH == 1) {
#pragma unroll
            for (int i = 0; i < 2; ++i) {
                const int idx = i * 256 + tid;
                const int row = idx >> 2, c8 = (idx & 3) * 8;
                *reinterpret_cast<bf16x8*>(&Bs[buf][row][c8]) = brh[i];
            }
        } else if constexpr (BPATH == 2) {
#pragma unroll
            for (int i = 0; i < 4; ++i) {
                const int idx = i * 256 + tid;
                const int kr = idx >> 5, cb = (idx & 31) * 4;
                Bs[buf][cb + 0][kr] = (bf16)brf[i].x;
                Bs[buf][cb + 1][kr] = (bf16)brf[i].y;
                Bs[buf][cb + 2][kr] = (bf16)brf[i].z;
                Bs[buf][cb + 3][kr] = (bf16)brf[i].w;
            }
        }
    };

#if HAVE_TDM
    auto tdm_B = [&](int kt, int buf) {
        if constexpr (BPATH == 0) {
            if (wave == 0) {
                const unsigned lds_off = (unsigned)(size_t)(&Bs[buf][0][0]);
                tdm_load_B(Wt + (size_t)gn0 * K_TOTAL + kt * 32, lds_off);
            }
        }
    };
#endif

    auto compute = [&](int buf) {
        // ISA 7.12.2 bf16 A 16x32: lanes 0-15 (M=l)   k = {0..7, 16..23}
        //                          lanes 16-31 (M=l') k = {8..15, 24..31}
        const int l15   = lane & 15;
        const int koffA = (lane >> 4) * 8;
        const int koffB = (lane >> 4) * 8;    // B 32x16: lanes 0-15 k={0..7,16..23}, hi {8..15,24..31}
        v16bf a[4], b[2];
#pragma unroll
        for (int mt = 0; mt < 4; ++mt) {
            const bf16* p = &As[buf][wm + mt * 16 + l15][koffA];
            bf16x8 lo = *reinterpret_cast<const bf16x8*>(p);
            bf16x8 hi = *reinterpret_cast<const bf16x8*>(p + 16);
            a[mt] = __builtin_shufflevector(lo, hi, 0,1,2,3,4,5,6,7,8,9,10,11,12,13,14,15);
        }
#pragma unroll
        for (int nt = 0; nt < 2; ++nt) {
            const bf16* p = &Bs[buf][wn + nt * 16 + l15][koffB];
            bf16x8 lo = *reinterpret_cast<const bf16x8*>(p);
            bf16x8 hi = *reinterpret_cast<const bf16x8*>(p + 16);
            b[nt] = __builtin_shufflevector(lo, hi, 0,1,2,3,4,5,6,7,8,9,10,11,12,13,14,15);
        }
#pragma unroll
        for (int mt = 0; mt < 4; ++mt)
#pragma unroll
            for (int nt = 0; nt < 2; ++nt)
                acc[mt][nt] = __builtin_amdgcn_wmma_f32_16x16x32_bf16(
                    false, a[mt], false, b[nt], (short)0, acc[mt][nt], false, false);
    };

    // ---- prologue: fill buffer 0
#if HAVE_TDM
    tdm_B(0, 0);
#endif
    load_A(0);  load_B(0);
    store_A(0); store_B(0);
#if HAVE_TDM
    if constexpr (BPATH == 0)
        if (wave == 0) __builtin_amdgcn_s_wait_tensorcnt((short)0);
#endif
    __syncthreads();

    for (int kt = 0; kt < KT_STEPS; ++kt) {
        const int cur = kt & 1;
        const bool more = (kt + 1 < KT_STEPS);
        if (more) {
#if HAVE_TDM
            tdm_B(kt + 1, 1 - cur);        // DMA runs while we compute
#endif
            load_A(kt + 1);
            load_B(kt + 1);
        }
        if (kt + 2 < KT_STEPS) {           // L2 prefetch 2 tiles ahead (A stream)
            const int kg = (kt + 2) * 32;
            const float* asrc = (kg < 2048) ? attn1 : attn2;
            __builtin_prefetch(asrc + (size_t)(gm0 + (tid >> 3)) * D_MODEL + (kg & 2047), 0, 1);
        }
        compute(cur);
        if (more) {
            store_A(1 - cur);
            store_B(1 - cur);
#if HAVE_TDM
            if constexpr (BPATH == 0)
                if (wave == 0) __builtin_amdgcn_s_wait_tensorcnt((short)0);
#endif
        }
        __syncthreads();
    }

    // ---- epilogue: +bias, store x_s. C layout: VGPR r -> M=r / M=8+r.
#pragma unroll
    for (int nt = 0; nt < 2; ++nt) {
        const int col = gn0 + wn + nt * 16 + (lane & 15);
        const float bias = bsh[col];
#pragma unroll
        for (int mt = 0; mt < 4; ++mt) {
            const int rbase = gm0 + wm + mt * 16 + ((lane >> 4) * 8);
#pragma unroll
            for (int r = 0; r < 8; ++r)
                xs_out[(size_t)(rbase + r) * D_MODEL + col] = acc[mt][nt][r] + bias;
        }
    }
}

// ---------------------------------------------------------------------------
// threefry2x32 with 16 rounds, key = jax.random.key(42) -> (0,42). random_bits over
// n=49152 counters: pair (i, i+24576) -> outputs concat(o0, o1).
// ---------------------------------------------------------------------------
__device__ __forceinline__ unsigned rotl32(unsigned x, int r) {
    return (x << r) | (x >> (32 - r));
}

__device__ __forceinline__ void threefry2x32(unsigned k0, unsigned k1,
                                             unsigned x0, unsigned x1,
                                             unsigned& o0, unsigned& o1)
{
    const unsigned ks2 = 0x1BD11BDAu ^ k0 ^ k1;
    const int R0[4] = {13, 15, 26, 6};
    const int R1[4] = {17, 29, 16, 24};
    x0 += k0; x1 += k1;
#pragma unroll
    for (int i = 0; i < 4; ++i) { x0 += x1; x1 = rotl32(x1, R0[i]); x1 ^= x0; }
    x0 += k1; x1 += ks2 + 1u;
#pragma unroll
    for (int i = 0; i < 4; ++i) { x0 += x1; x1 = rotl32(x1, R1[i]); x1 ^= x0; }
    x0 += ks2; x1 += k0 + 2u;
#pragma unroll
    for (int i = 0; i < 4; ++i) { x0 += x1; x1 = rotl32(x1, R0[i]); x1 ^= x0; }
    x0 += k0; x1 += k1 + 3u;
#pragma unroll
    for (int i = 0; i < 4; ++i) { x0 += x1; x1 = rotl32(x1, R1[i]); x1 ^= x0; }
    x0 += k1; x1 += ks2 + 4u;
    o0 = x0; o1 = x1;
}

__device__ __forceinline__ float gumbel_for(int flat)  // flat in [0, 49152)
{
    const unsigned HALF = 24576u;
    const unsigned c = (unsigned)flat % HALF;
    unsigned o0, o1;
    threefry2x32(0u, 42u, c, c + HALF, o0, o1);
    const unsigned bits = ((unsigned)flat < HALF) ? o0 : o1;
    float U = __uint_as_float((bits >> 9) | 0x3f800000u) - 1.0f;  // [0,1)
    U = fminf(fmaxf(U, 1e-10f), 1.0f);
    return -logf(-logf(U));
}

// ---------------------------------------------------------------------------
// Kernel 2: router + hard selection. One wave32 per token.
// ---------------------------------------------------------------------------
__global__ __launch_bounds__(256)
void router_select_kernel(const float* __restrict__ attn1,
                          const float* __restrict__ attn2,
                          const float* __restrict__ Wr,   // [2048, 3]
                          const float* __restrict__ brt,  // [3]
                          float* __restrict__ out,        // holds x_s
                          unsigned* __restrict__ counts)  // [3]
{
    const int lane  = threadIdx.x & 31;
    const int wave  = threadIdx.x >> 5;
    const int token = blockIdx.x * 8 + wave;
    const float* xrow = out + (size_t)token * D_MODEL;

    float s0 = 0.f, s1 = 0.f, s2 = 0.f;
    for (int d = lane; d < D_MODEL; d += 32) {
        const float x = xrow[d];
        s0 = fmaf(x, Wr[d * 3 + 0], s0);
        s1 = fmaf(x, Wr[d * 3 + 1], s1);
        s2 = fmaf(x, Wr[d * 3 + 2], s2);
    }
#pragma unroll
    for (int off = 16; off > 0; off >>= 1) {   // wave32 butterfly, all lanes get sums
        s0 += __shfl_xor(s0, off, 32);
        s1 += __shfl_xor(s1, off, 32);
        s2 += __shfl_xor(s2, off, 32);
    }

    const float l0 = s0 + brt[0] + gumbel_for(token * 3 + 0);
    const float l1 = s1 + brt[1] + gumbel_for(token * 3 + 1);
    const float l2 = s2 + brt[2] + gumbel_for(token * 3 + 2);
    int idx = 0; float best = l0;
    if (l1 > best) { best = l1; idx = 1; }
    if (l2 > best) { best = l2; idx = 2; }

    if (idx != 2) {
        const float* src = (idx == 0) ? attn1 : attn2;
        const float4* sp = reinterpret_cast<const float4*>(src + (size_t)token * D_MODEL);
        float4* dp = reinterpret_cast<float4*>(out + (size_t)token * D_MODEL);
        for (int d = lane; d < D_MODEL / 4; d += 32) dp[d] = sp[d];
    }
    if (lane == 0) atomicAdd(&counts[idx], 1u);
}

// ---------------------------------------------------------------------------
__global__ void finalize_kernel(const unsigned* __restrict__ counts,
                                float* __restrict__ out_scalars)
{
    if (threadIdx.x == 0) {
        const float n = (float)N_TOKENS;
        float lbl = 0.f;
        for (int j = 0; j < 3; ++j) {
            const float mp = (float)counts[j] / n;
            lbl -= mp * logf(mp + 1e-8f);
        }
        out_scalars[0] = logf(1.0f + 1e-8f);  // entropy_loss (one-hot p)
        out_scalars[1] = lbl;                 // load_balance_loss
    }
}

__global__ void zero_counts_kernel(unsigned* __restrict__ c)
{
    if (threadIdx.x < 3) c[threadIdx.x] = 0u;
}

// ---------------------------------------------------------------------------
extern "C" void kernel_launch(void* const* d_in, const int* in_sizes, int n_in,
                              void* d_out, int out_size, void* d_ws, size_t ws_size,
                              hipStream_t stream)
{
    const float* attn1 = (const float*)d_in[0];
    const float* attn2 = (const float*)d_in[1];
    const float* Ws    = (const float*)d_in[2];
    const float* bsh   = (const float*)d_in[3];
    const float* Wr    = (const float*)d_in[4];
    const float* brt   = (const float*)d_in[5];
    float* out = (float*)d_out;

    unsigned* counts = (unsigned*)d_ws;
    bf16* Wt = (bf16*)((char*)d_ws + 256);
    const size_t wt_bytes = (size_t)D_MODEL * K_TOTAL * sizeof(bf16);   // 16 MB
    const bool have_wt = ws_size >= 256 + wt_bytes;

    zero_counts_kernel<<<1, 32, 0, stream>>>(counts);

    dim3 grid(D_MODEL / 128, N_TOKENS / 128);   // (16, 128)
    if (have_wt) {
        transpose_ws_kernel<<<dim3(K_TOTAL / 32, D_MODEL / 32), 256, 0, stream>>>(Ws, Wt);
#if HAVE_TDM
        gemm_xs_kernel<0><<<grid, 256, 0, stream>>>(attn1, attn2, Ws, Wt, bsh, out);
#else
        gemm_xs_kernel<1><<<grid, 256, 0, stream>>>(attn1, attn2, Ws, Wt, bsh, out);
#endif
    } else {
        gemm_xs_kernel<2><<<grid, 256, 0, stream>>>(attn1, attn2, Ws, nullptr, bsh, out);
    }

    router_select_kernel<<<N_TOKENS / 8, 256, 0, stream>>>(attn1, attn2, Wr, brt,
                                                           out, counts);
    finalize_kernel<<<1, 32, 0, stream>>>(counts, out + (size_t)N_TOKENS * D_MODEL);
}
